// MultiHeadWindowsAttention_21406117003585
// MI455X (gfx1250) — matrix operand order repaired and checked by reading, not compiled, not run on hardware
//
#include <hip/hip_runtime.h>
#include <hip/hip_bf16.h>

// ---------------------------------------------------------------------------
// MI455X / gfx1250 wave32 WMMA implementation of MultiHeadWindowsAttention.
// All GEMM-shaped work (convs, qkv/proj linears, attention) runs on
// v_wmma_f32_16x16x32_bf16 (bf16 operands, f32 accumulate).
// ---------------------------------------------------------------------------

typedef __bf16 v16bf __attribute__((ext_vector_type(16)));
typedef __bf16 v8bf  __attribute__((ext_vector_type(8)));
typedef float  v8f   __attribute__((ext_vector_type(8)));

__device__ __forceinline__ v16bf cat8(v8bf a, v8bf b) {
  return __builtin_shufflevector(a, b, 0,1,2,3,4,5,6,7,8,9,10,11,12,13,14,15);
}
// A fragment (16xK rows, K chunk = 32).  p = this lane's row at current K chunk
// (k-contiguous).  ISA layout: elems 0..7 -> k = kh*8+e ; 8..15 -> 16+kh*8+e.
__device__ __forceinline__ v16bf frag_a(const __bf16* p, int kh) {
  v8bf lo = *(const v8bf*)(p + kh * 8);
  v8bf hi = *(const v8bf*)(p + 16 + kh * 8);
  return cat8(lo, hi);
}
// B fragment (Kx16 cols, K chunk = 32).  p = this lane's column data
// (k-contiguous) at current K chunk.  ISA: elems 0..7 -> k = kh*8+e ; 8..15 -> 16+kh*8+e.
__device__ __forceinline__ v16bf frag_b(const __bf16* p, int kh) {
  v8bf lo = *(const v8bf*)(p + kh * 8);
  v8bf hi = *(const v8bf*)(p + 16 + kh * 8);
  return cat8(lo, hi);
}
__device__ __forceinline__ v8f wmma_bf16(v16bf a, v16bf b, v8f c) {
  return __builtin_amdgcn_wmma_f32_16x16x32_bf16(false, a, false, b,
                                                 (short)0, c, false, false);
}

// ---------------------------------------------------------------------------
// Weight packing: conv OIHW f32 -> [ky*3+kx][co][ci] bf16  (ci contiguous)
// ---------------------------------------------------------------------------
__global__ void pack_conv_w_kernel(const float* __restrict__ w,
                                   __bf16* __restrict__ out, int OC, int IC) {
  int i = blockIdx.x * 256 + threadIdx.x;
  int n = OC * IC * 9;
  if (i >= n) return;
  int kk = i % 9;
  int ci = (i / 9) % IC;
  int co = i / (9 * IC);
  out[((size_t)kk * OC + co) * IC + ci] = (__bf16)w[i];
}

__global__ void pack_f32_bf16_kernel(const float* __restrict__ w,
                                     __bf16* __restrict__ out, int n) {
  int i = blockIdx.x * 256 + threadIdx.x;
  if (i < n) out[i] = (__bf16)w[i];
}

// ---------------------------------------------------------------------------
// 3x3 SAME conv as implicit GEMM.  Orientation: M = out-channel, N = pixel,
// K = in-channel -> C/D columns are pixels => coalesced stores.
// Block = 256 thr (8 waves): tile 64 px (one row) x 32 co.
//   wave: cot = wave&1 (16 co), pxt = wave>>1 (16 px).
// Activations staged per (ky, ci-chunk) in LDS [66 px][40 ci] (pad vs banks).
// ---------------------------------------------------------------------------
template <typename Tin, bool OUT_BF16, bool ADD_IDENT>
__global__ __launch_bounds__(256)
void conv3x3_wmma_kernel(const Tin* __restrict__ in,
                         const __bf16* __restrict__ wpk,
                         const float* __restrict__ bias,
                         void* __restrict__ outp,
                         const float* __restrict__ ident,
                         int B, int Cin, int OC, int H, int W) {
  __shared__ __bf16 atile[66][40];

  const int tid  = threadIdx.x;
  const int lane = tid & 31;
  const int wave = tid >> 5;
  const int lan15 = lane & 15;
  const int kh    = lane >> 4;

  const int x0 = blockIdx.x * 64;
  const int y  = blockIdx.y;
  const int nCblk = OC >> 5;
  const int b   = blockIdx.z / nCblk;
  const int co0 = (blockIdx.z % nCblk) << 5;

  const int cot = wave & 1;
  const int pxt = wave >> 1;

  const size_t HW = (size_t)H * W;
  v8f acc = (v8f)0.0f;

  const int nK = Cin >> 5;
  for (int ky = 0; ky < 3; ++ky) {
    const int yy = y + ky - 1;
    if (yy < 0 || yy >= H) continue;  // uniform branch (y uniform per block)
    for (int cc = 0; cc < nK; ++cc) {
      __syncthreads();
      {  // cooperative load: 66 px x 32 ci, coalesced along x
        const size_t baseo = ((size_t)(b * Cin + (cc << 5)) * H + yy) * (size_t)W;
        for (int i = tid; i < 66 * 32; i += 256) {
          int px = i % 66;
          int c  = i / 66;
          int xg = x0 - 1 + px;
          float v = 0.0f;
          if (xg >= 0 && xg < W) v = (float)in[baseo + (size_t)c * HW + xg];
          atile[px][c] = (__bf16)v;
        }
      }
      __syncthreads();
      for (int kx = 0; kx < 3; ++kx) {
        const int co_r = co0 + (cot << 4) + lan15;  // A row (lane&15)
        const __bf16* ap =
            wpk + ((size_t)(ky * 3 + kx) * OC + co_r) * Cin + (cc << 5);
        v16bf afrag = frag_a(ap, kh);
        const int pxi = (pxt << 4) + lan15 + kx;    // B col (lane&15), +kx shift
        v16bf bfrag = frag_b(&atile[pxi][0], kh);
        acc = wmma_bf16(afrag, bfrag, acc);
      }
    }
  }

  // Epilogue: C element r -> row co = base + r + 8*kh, col x = base + lane&15
  const int xo = x0 + (pxt << 4) + lan15;
#pragma unroll
  for (int r = 0; r < 8; ++r) {
    int co = co0 + (cot << 4) + r + (kh << 3);
    float val = acc[r] + bias[co];
    size_t off = ((size_t)(b * OC + co) * H + y) * (size_t)W + xo;
    if (ADD_IDENT) val += ident[off];
    if (OUT_BF16)
      ((__bf16*)outp)[off] = (__bf16)val;
    else
      ((float*)outp)[off] = val;
  }
}

// ---------------------------------------------------------------------------
// Fused window attention: one block (8 waves) per 8x8 window (64 tok x 256 C).
//   phase0: stage wm (f32->bf16), wf (bf16) token-major, pitch 264
//   phase1: q_g = sigmoid(wf@Wqf^T+b)*(wm@Wqm^T+b), k_g likewise, v = wm@Wv^T
//           (v stored transposed vt[c][tok] for the out GEMM)
//   phase2: per (head, 16-row tile): S = q_g k_g^T * 1/sqrt(32), row softmax
//           via __shfl_xor (wave32), P restaged per-wave in LDS, out = P @ V
//   phase3: proj GEMM -> attn_out bf16 NCHW (window_reverse folded in)
// LDS reuse: outb aliases wm, P aliases wf.  Peak = 168 KB (< 320 KB / WGP).
// ---------------------------------------------------------------------------
#define CPITCH 264
#define VTP 72
#define PPITCH 72
#define ATTN_LDS_BYTES 172032

__global__ __launch_bounds__(256)
void win_attn_kernel(const float* __restrict__ xm,    // conv1 out, f32 NCHW
                     const __bf16* __restrict__ xf,   // conv2 out, bf16 NCHW
                     const __bf16* __restrict__ Wqm, const float* __restrict__ bqm,
                     const __bf16* __restrict__ Wkm, const float* __restrict__ bkm,
                     const __bf16* __restrict__ Wvm, const float* __restrict__ bvm,
                     const __bf16* __restrict__ Wqf, const float* __restrict__ bqf,
                     const __bf16* __restrict__ Wkf, const float* __restrict__ bkf,
                     const __bf16* __restrict__ Wpr, const float* __restrict__ bpr,
                     __bf16* __restrict__ attn_out,
                     int H, int W) {
  extern __shared__ char smem[];
  __bf16* wm   = (__bf16*)(smem + 0);        // 64*264*2 = 33792
  __bf16* wf   = (__bf16*)(smem + 33792);
  __bf16* qg   = (__bf16*)(smem + 67584);
  __bf16* kg   = (__bf16*)(smem + 101376);
  __bf16* vt   = (__bf16*)(smem + 135168);   // 256*72*2 = 36864 (end 172032)
  __bf16* outb = (__bf16*)(smem + 0);        // alias wm (dead after phase1)
  __bf16* pbuf = (__bf16*)(smem + 33792);    // alias wf (dead after phase1)

  const int tid = threadIdx.x, lane = tid & 31, wave = tid >> 5;
  const int lan15 = lane & 15, kh = lane >> 4;

  const int nWx = W >> 3, nWy = H >> 3;
  const int wi  = blockIdx.x;
  const int b   = wi / (nWx * nWy);
  const int rem = wi % (nWx * nWy);
  const int wy  = rem / nWx, wx = rem % nWx;
  const size_t HW = (size_t)H * W;
  const size_t gbase = (size_t)b * 256 * HW + (size_t)(wy * 8) * W + (wx * 8);

  // ---- phase 0: stage window tiles ----------------------------------------
  for (int i = tid; i < 64 * 256; i += 256) {
    int c = i >> 6, t = i & 63;
    int ty = t >> 3, tx = t & 7;
    size_t g = gbase + (size_t)c * HW + (size_t)ty * W + tx;
    wm[t * CPITCH + c] = (__bf16)xm[g];
    wf[t * CPITCH + c] = xf[g];
  }
  __syncthreads();

  // ---- phase 1: gated q/k + v (M=co, N=tok, K=ci) -------------------------
  for (int tt = wave; tt < 64; tt += 8) {      // q_g
    const int cot = tt & 15, ntk = tt >> 4;
    const int co_r = (cot << 4) + lan15;
    const int tok  = (ntk << 4) + lan15;
    v8f am = (v8f)0.0f, af = (v8f)0.0f;
    for (int cc = 0; cc < 8; ++cc) {
      const int kb = cc << 5;
      v16bf bm = frag_b(wm + tok * CPITCH + kb, kh);
      v16bf bff = frag_b(wf + tok * CPITCH + kb, kh);
      am = wmma_bf16(frag_a(Wqm + (size_t)co_r * 256 + kb, kh), bm, am);
      af = wmma_bf16(frag_a(Wqf + (size_t)co_r * 256 + kb, kh), bff, af);
    }
#pragma unroll
    for (int r = 0; r < 8; ++r) {
      int co = (cot << 4) + r + (kh << 3);
      float m = am[r] + bqm[co];
      float f = af[r] + bqf[co];
      qg[tok * CPITCH + co] = (__bf16)(m * (1.0f / (1.0f + __expf(-f))));
    }
  }
  for (int tt = wave; tt < 64; tt += 8) {      // k_g
    const int cot = tt & 15, ntk = tt >> 4;
    const int co_r = (cot << 4) + lan15;
    const int tok  = (ntk << 4) + lan15;
    v8f am = (v8f)0.0f, af = (v8f)0.0f;
    for (int cc = 0; cc < 8; ++cc) {
      const int kb = cc << 5;
      v16bf bm = frag_b(wm + tok * CPITCH + kb, kh);
      v16bf bff = frag_b(wf + tok * CPITCH + kb, kh);
      am = wmma_bf16(frag_a(Wkm + (size_t)co_r * 256 + kb, kh), bm, am);
      af = wmma_bf16(frag_a(Wkf + (size_t)co_r * 256 + kb, kh), bff, af);
    }
#pragma unroll
    for (int r = 0; r < 8; ++r) {
      int co = (cot << 4) + r + (kh << 3);
      float m = am[r] + bkm[co];
      float f = af[r] + bkf[co];
      kg[tok * CPITCH + co] = (__bf16)(m * (1.0f / (1.0f + __expf(-f))));
    }
  }
  for (int tt = wave; tt < 64; tt += 8) {      // v (stored transposed)
    const int cot = tt & 15, ntk = tt >> 4;
    const int co_r = (cot << 4) + lan15;
    const int tok  = (ntk << 4) + lan15;
    v8f av = (v8f)0.0f;
    for (int cc = 0; cc < 8; ++cc) {
      const int kb = cc << 5;
      av = wmma_bf16(frag_a(Wvm + (size_t)co_r * 256 + kb, kh),
                     frag_b(wm + tok * CPITCH + kb, kh), av);
    }
#pragma unroll
    for (int r = 0; r < 8; ++r) {
      int co = (cot << 4) + r + (kh << 3);
      vt[co * VTP + tok] = (__bf16)(av[r] + bvm[co]);  // contiguous over tok
    }
  }
  __syncthreads();

  // ---- phase 2: attention per (head, row tile) ----------------------------
  const float scale = 0.17677669529663687f;  // 1/sqrt(32)
  __bf16* pw = pbuf + wave * 16 * PPITCH;    // per-wave P tile (16 x 64)
  for (int task = wave; task < 32; task += 8) {
    const int h = task >> 2, ntk = task & 3;
    const int hc = h << 5;
    // scores S[16 n x 64 m]:  M=n, N=m, K=d(32)
    v16bf aq = frag_a(qg + ((ntk << 4) + lan15) * CPITCH + hc, kh);
    v8f s[4];
#pragma unroll
    for (int mt = 0; mt < 4; ++mt) {
      s[mt] = (v8f)0.0f;
      v16bf bk = frag_b(kg + ((mt << 4) + lan15) * CPITCH + hc, kh);
      s[mt] = wmma_bf16(aq, bk, s[mt]);
    }
    // row softmax over m (16 lanes in group x 4 tiles)
#pragma unroll
    for (int r = 0; r < 8; ++r) {
      float e0 = s[0][r] * scale, e1 = s[1][r] * scale;
      float e2 = s[2][r] * scale, e3 = s[3][r] * scale;
      float mx = fmaxf(fmaxf(e0, e1), fmaxf(e2, e3));
      mx = fmaxf(mx, __shfl_xor(mx, 1, 32));
      mx = fmaxf(mx, __shfl_xor(mx, 2, 32));
      mx = fmaxf(mx, __shfl_xor(mx, 4, 32));
      mx = fmaxf(mx, __shfl_xor(mx, 8, 32));
      e0 = __expf(e0 - mx); e1 = __expf(e1 - mx);
      e2 = __expf(e2 - mx); e3 = __expf(e3 - mx);
      float sum = e0 + e1 + e2 + e3;
      sum += __shfl_xor(sum, 1, 32);
      sum += __shfl_xor(sum, 2, 32);
      sum += __shfl_xor(sum, 4, 32);
      sum += __shfl_xor(sum, 8, 32);
      float inv = 1.0f / sum;
      int nl = r + (kh << 3);                 // C row -> local token
      pw[nl * PPITCH +  0 + lan15] = (__bf16)(e0 * inv);
      pw[nl * PPITCH + 16 + lan15] = (__bf16)(e1 * inv);
      pw[nl * PPITCH + 32 + lan15] = (__bf16)(e2 * inv);
      pw[nl * PPITCH + 48 + lan15] = (__bf16)(e3 * inv);
    }
    // out = P @ V : M=n(16), N=d(16 x2), K=m(64)
#pragma unroll
    for (int dt = 0; dt < 2; ++dt) {
      v8f o = (v8f)0.0f;
#pragma unroll
      for (int ks = 0; ks < 2; ++ks) {
        v16bf ap = frag_a(pw + lan15 * PPITCH + (ks << 5), kh);
        v16bf bv = frag_b(vt + (hc + (dt << 4) + lan15) * VTP + (ks << 5), kh);
        o = wmma_bf16(ap, bv, o);
      }
#pragma unroll
      for (int r = 0; r < 8; ++r) {
        int tok = (ntk << 4) + r + (kh << 3);
        int c   = hc + (dt << 4) + lan15;
        outb[tok * CPITCH + c] = (__bf16)o[r];
      }
    }
  }
  __syncthreads();

  // ---- phase 3: proj + window_reverse store -------------------------------
  for (int tt = wave; tt < 64; tt += 8) {
    const int cot = tt & 15, ntk = tt >> 4;
    const int co_r = (cot << 4) + lan15;
    const int tok  = (ntk << 4) + lan15;
    v8f ac = (v8f)0.0f;
    for (int cc = 0; cc < 8; ++cc) {
      const int kb = cc << 5;
      ac = wmma_bf16(frag_a(Wpr + (size_t)co_r * 256 + kb, kh),
                     frag_b(outb + tok * CPITCH + kb, kh), ac);
    }
#pragma unroll
    for (int r = 0; r < 8; ++r) {
      int co = (cot << 4) + r + (kh << 3);
      float v = ac[r] + bpr[co];
      int t  = (ntk << 4) + lan15;            // C col -> token
      int ty = t >> 3, tx = t & 7;
      size_t off = (size_t)(b * 256 + co) * HW +
                   (size_t)(wy * 8 + ty) * W + (wx * 8 + tx);
      attn_out[off] = (__bf16)v;
    }
  }
}

// ---------------------------------------------------------------------------
extern "C" void kernel_launch(void* const* d_in, const int* in_sizes, int n_in,
                              void* d_out, int out_size, void* d_ws,
                              size_t ws_size, hipStream_t stream) {
  const int B = 4, C = 256, H = 256, W = 256, CF = 32;

  const float* xmap    = (const float*)d_in[0];
  const float* xfeat   = (const float*)d_in[1];
  const float* cmap_w  = (const float*)d_in[2];
  const float* cmap_b  = (const float*)d_in[3];
  const float* cfeat_w = (const float*)d_in[4];
  const float* cfeat_b = (const float*)d_in[5];
  const float* qxmap_w = (const float*)d_in[6];
  const float* qxmap_b = (const float*)d_in[7];
  const float* kxmap_w = (const float*)d_in[8];
  const float* kxmap_b = (const float*)d_in[9];
  const float* vxmap_w = (const float*)d_in[10];
  const float* vxmap_b = (const float*)d_in[11];
  const float* qxfeat_w = (const float*)d_in[12];
  const float* qxfeat_b = (const float*)d_in[13];
  const float* kxfeat_w = (const float*)d_in[14];
  const float* kxfeat_b = (const float*)d_in[15];
  const float* proj_w  = (const float*)d_in[16];
  const float* proj_b  = (const float*)d_in[17];
  const float* cout_w  = (const float*)d_in[18];
  const float* cout_b  = (const float*)d_in[19];

  // workspace carve-out (256 B aligned)
  char* ws = (char*)d_ws;
  size_t off = 0;
  auto carve = [&](size_t bytes) -> void* {
    off = (off + 255) & ~(size_t)255;
    void* p = ws + off;
    off += bytes;
    return p;
  };
  const size_t NPIX = (size_t)B * C * H * W;           // 67.1M elements
  float*  conv1o = (float*)carve(NPIX * 4);            // f32 (also identity)
  __bf16* conv2o = (__bf16*)carve(NPIX * 2);
  __bf16* attno  = (__bf16*)carve(NPIX * 2);
  __bf16* Wc1 = (__bf16*)carve((size_t)9 * C * C * 2);
  __bf16* Wc2 = (__bf16*)carve((size_t)9 * C * CF * 2);
  __bf16* Wc3 = (__bf16*)carve((size_t)9 * C * C * 2);
  __bf16* Wqm = (__bf16*)carve((size_t)C * C * 2);
  __bf16* Wkm = (__bf16*)carve((size_t)C * C * 2);
  __bf16* Wvm = (__bf16*)carve((size_t)C * C * 2);
  __bf16* Wqf = (__bf16*)carve((size_t)C * C * 2);
  __bf16* Wkf = (__bf16*)carve((size_t)C * C * 2);
  __bf16* Wpr = (__bf16*)carve((size_t)C * C * 2);
  if (off > ws_size) return;  // workspace too small: nothing safe to do

  // pack weights to bf16, GEMM-friendly layouts
  {
    int n1 = C * C * 9;
    pack_conv_w_kernel<<<(n1 + 255) / 256, 256, 0, stream>>>(cmap_w, Wc1, C, C);
    int n2 = C * CF * 9;
    pack_conv_w_kernel<<<(n2 + 255) / 256, 256, 0, stream>>>(cfeat_w, Wc2, C, CF);
    pack_conv_w_kernel<<<(n1 + 255) / 256, 256, 0, stream>>>(cout_w, Wc3, C, C);
    int nl = C * C;
    pack_f32_bf16_kernel<<<(nl + 255) / 256, 256, 0, stream>>>(qxmap_w, Wqm, nl);
    pack_f32_bf16_kernel<<<(nl + 255) / 256, 256, 0, stream>>>(kxmap_w, Wkm, nl);
    pack_f32_bf16_kernel<<<(nl + 255) / 256, 256, 0, stream>>>(vxmap_w, Wvm, nl);
    pack_f32_bf16_kernel<<<(nl + 255) / 256, 256, 0, stream>>>(qxfeat_w, Wqf, nl);
    pack_f32_bf16_kernel<<<(nl + 255) / 256, 256, 0, stream>>>(kxfeat_w, Wkf, nl);
    pack_f32_bf16_kernel<<<(nl + 255) / 256, 256, 0, stream>>>(proj_w, Wpr, nl);
  }

  dim3 cgrid(W / 64, H, B * (C / 32));

  // conv1: xmap f32 -> conv1o f32 (identity kept at full precision)
  conv3x3_wmma_kernel<float, false, false><<<cgrid, 256, 0, stream>>>(
      xmap, Wc1, cmap_b, (void*)conv1o, nullptr, B, C, C, H, W);

  // conv2: xfeat f32 (32 ch) -> conv2o bf16
  conv3x3_wmma_kernel<float, true, false><<<cgrid, 256, 0, stream>>>(
      xfeat, Wc2, cfeat_b, (void*)conv2o, nullptr, B, CF, C, H, W);

  // fused window attention -> attno bf16
  int nwin = B * (H / 8) * (W / 8);
  win_attn_kernel<<<nwin, 256, ATTN_LDS_BYTES, stream>>>(
      conv1o, conv2o, Wqm, qxmap_b, Wkm, kxmap_b, Wvm, vxmap_b, Wqf, qxfeat_b,
      Wkf, kxfeat_b, Wpr, proj_b, attno, H, W);

  // conv3: attno bf16 -> f32 out, + identity (conv1o)
  conv3x3_wmma_kernel<__bf16, false, true><<<cgrid, 256, 0, stream>>>(
      attno, Wc3, cout_b, d_out, conv1o, B, C, C, H, W);
}
